// LeagueAwareModel_36094905155719
// MI455X (gfx1250) — matrix operand order repaired and checked by reading, not compiled, not run on hardware
//
#include <hip/hip_runtime.h>
#include <hip/hip_bf16.h>
#include <math.h>

// ---------------------------------------------------------------------------
// Model constants
// ---------------------------------------------------------------------------
#define BATCH      1024
#define DIM        256
#define LDIM       32
#define HEADS      8
#define SEQ        10
#define NTEAMS     8192
#define FUSION     2112

// d_out layout (floats): lambdas(2048) rho(1024) xg(2048) h_s(262144) a_s(262144)
#define OUT_LAMBDA 0
#define OUT_RHO    2048
#define OUT_XG     3072
#define OUT_HS     5120
#define OUT_AS     (5120 + 1024*256)

// ---------------------------------------------------------------------------
// bf16 helpers + WMMA vector types
// ---------------------------------------------------------------------------
typedef __bf16 bf16_t;
typedef __attribute__((ext_vector_type(2)))  __bf16 v2bf;
typedef __attribute__((ext_vector_type(8)))  __bf16 v8bf;
typedef __attribute__((ext_vector_type(16))) __bf16 v16bf;
typedef __attribute__((ext_vector_type(8)))  float  v8f;

// packed f32x2 -> bf16x2: prefer the amdgcn builtin; otherwise let the BACKEND
// lower fptrunc (native v_cvt_*_bf16_f32 on targets that have it, else the
// standard RNE expansion).
__device__ __forceinline__ v2bf pk2(float x, float y) {
#if __has_builtin(__builtin_amdgcn_cvt_pk_bf16_f32)
    return __builtin_amdgcn_cvt_pk_bf16_f32(x, y);
#else
    v2bf r; r[0] = (bf16_t)x; r[1] = (bf16_t)y; return r;
#endif
}

__device__ __forceinline__ v16bf cat16(v8bf lo, v8bf hi) {
    v16bf r;
#pragma unroll
    for (int i = 0; i < 8; ++i) { r[i] = lo[i]; r[i + 8] = hi[i]; }
    return r;
}

// ---------------------------------------------------------------------------
// WMMA GEMM:  C[M,N] = act(A[M,K] @ W[K,N] + bias)
// requirements (all call sites satisfy them): M%128==0, N%128==0, K%32==0
// block: 256 threads = 8 waves (4x2); block tile 128x128; wave tile 32x64
// => 8 x v_wmma_f32_16x16x32_bf16 per K-step, double-buffered LDS,
//    ONE barrier per K-step, global loads for tile k+1 overlap WMMAs of tile k
// ---------------------------------------------------------------------------
enum { ACT_NONE = 0, ACT_RELU = 1, ACT_TANH = 2, ACT_LEAKY01 = 3 };

template <int ACT>
__global__ __launch_bounds__(256) void gemm_wmma(
    const float* __restrict__ A, const float* __restrict__ W,
    const float* __restrict__ bias, float* __restrict__ C,
    int M, int N, int Kdim)
{
    // double-buffered tiles, bf16, padded stride 40 halves (80B, 16B-aligned rows)
    __shared__ __align__(16) __bf16 lA[2][128 * 40];   // [m][k]
    __shared__ __align__(16) __bf16 lB[2][128 * 40];   // W tile TRANSPOSED [n][k]

    const int tid   = threadIdx.x;
    const int wave  = tid >> 5;
    const int lane  = tid & 31;
    const int waveM = wave >> 1;            // 0..3  (rows, 32 each)
    const int waveN = wave & 1;             // 0..1  (cols, 64 each)
    const int blockM = blockIdx.y * 128;
    const int blockN = blockIdx.x * 128;
    const int half = lane >> 4;             // 0/1
    const int l16  = lane & 15;

    // per-thread load slots: A 128x32 -> 4 float4 ; B 32x128 -> 4 float4
    const int aR = (tid * 4) >> 5,  aC = (tid * 4) & 31;    // + p*32 rows
    const int bK = (tid * 4) >> 7,  bC = (tid * 4) & 127;   // + p*8  k-rows

    v8f acc[2][4] = {};
    const int ktiles = Kdim / 32;

    // ---- prologue: stage tile 0 into lds buffer 0 ----
    {
        float4 ar[4], br[4];
#pragma unroll
        for (int p = 0; p < 4; ++p) {
            ar[p] = *reinterpret_cast<const float4*>(A + (size_t)(blockM + aR + p * 32) * Kdim + aC);
            br[p] = *reinterpret_cast<const float4*>(W + (size_t)(bK + p * 8) * N + blockN + bC);
        }
#pragma unroll
        for (int p = 0; p < 4; ++p) {
            __bf16* dst = &lA[0][(aR + p * 32) * 40 + aC];              // 8B-aligned (aC%4==0)
            *reinterpret_cast<v2bf*>(dst)     = pk2(ar[p].x, ar[p].y);
            *reinterpret_cast<v2bf*>(dst + 2) = pk2(ar[p].z, ar[p].w);
            int kr = bK + p * 8;
            v2bf b01 = pk2(br[p].x, br[p].y), b23 = pk2(br[p].z, br[p].w);
            lB[0][(bC + 0) * 40 + kr] = b01[0];
            lB[0][(bC + 1) * 40 + kr] = b01[1];
            lB[0][(bC + 2) * 40 + kr] = b23[0];
            lB[0][(bC + 3) * 40 + kr] = b23[1];
        }
    }
    __syncthreads();

    for (int kt = 0; kt < ktiles; ++kt) {
        const int cur = kt & 1, nxt = cur ^ 1;
        const int k1 = (kt + 1) * 32;

        // ---- issue global loads for tile kt+1 (overlap with WMMAs below) ----
        float4 ar[4], br[4];
        if (kt + 1 < ktiles) {
#pragma unroll
            for (int p = 0; p < 4; ++p) {
                ar[p] = *reinterpret_cast<const float4*>(A + (size_t)(blockM + aR + p * 32) * Kdim + k1 + aC);
                br[p] = *reinterpret_cast<const float4*>(W + (size_t)(k1 + bK + p * 8) * N + blockN + bC);
            }
            if (kt + 2 < ktiles) {   // keep the L2 ahead of us (global_prefetch_b8)
                __builtin_prefetch(A + (size_t)(blockM + (tid >> 1)) * Kdim + k1 + 32 + (tid & 1) * 16, 0, 1);
                __builtin_prefetch(W + (size_t)(k1 + 32 + (tid >> 3)) * N + blockN + ((tid & 7) << 4), 0, 1);
            }
        }

        // ---- fragments from lds[cur] ----
        // A 16x32 bf16 layout: lane m=l16, contiguous halves at K=half*8 and 16+half*8
        v16bf afrag[2], bfrag[4];
#pragma unroll
        for (int i = 0; i < 2; ++i) {
            int row = waveM * 32 + i * 16 + l16;
            v8bf lo = *reinterpret_cast<v8bf*>(&lA[cur][row * 40 + half * 8]);
            v8bf hi = *reinterpret_cast<v8bf*>(&lA[cur][row * 40 + 16 + half * 8]);
            afrag[i] = cat16(lo, hi);
        }
        // B 32x16: lane n=l16, contiguous halves at K=half*8 and 16+half*8
#pragma unroll
        for (int j = 0; j < 4; ++j) {
            int col = waveN * 64 + j * 16 + l16;
            v8bf lo = *reinterpret_cast<v8bf*>(&lB[cur][col * 40 + half * 8]);
            v8bf hi = *reinterpret_cast<v8bf*>(&lB[cur][col * 40 + 16 + half * 8]);
            bfrag[j] = cat16(lo, hi);
        }
#pragma unroll
        for (int i = 0; i < 2; ++i)
#pragma unroll
            for (int j = 0; j < 4; ++j)
                acc[i][j] = __builtin_amdgcn_wmma_f32_16x16x32_bf16(
                    false, afrag[i], false, bfrag[j], (short)0, acc[i][j], false, false);

        // ---- convert & park tile kt+1 into lds[nxt] ----
        if (kt + 1 < ktiles) {
#pragma unroll
            for (int p = 0; p < 4; ++p) {
                __bf16* dst = &lA[nxt][(aR + p * 32) * 40 + aC];
                *reinterpret_cast<v2bf*>(dst)     = pk2(ar[p].x, ar[p].y);
                *reinterpret_cast<v2bf*>(dst + 2) = pk2(ar[p].z, ar[p].w);
                int kr = bK + p * 8;
                v2bf b01 = pk2(br[p].x, br[p].y), b23 = pk2(br[p].z, br[p].w);
                lB[nxt][(bC + 0) * 40 + kr] = b01[0];
                lB[nxt][(bC + 1) * 40 + kr] = b01[1];
                lB[nxt][(bC + 2) * 40 + kr] = b23[0];
                lB[nxt][(bC + 3) * 40 + kr] = b23[1];
            }
        }
        __syncthreads();    // single barrier per K-step (double buffer makes it safe)
    }

    // ---- epilogue: C/D layout -> lane col=l16, rows v + half*8 ----
#pragma unroll
    for (int i = 0; i < 2; ++i) {
#pragma unroll
        for (int j = 0; j < 4; ++j) {
            int col = blockN + waveN * 64 + j * 16 + l16;
            float bv = bias ? bias[col] : 0.f;
#pragma unroll
            for (int v = 0; v < 8; ++v) {
                int row = blockM + waveM * 32 + i * 16 + v + half * 8;
                float x = acc[i][j][v] + bv;
                if (ACT == ACT_RELU)    x = fmaxf(x, 0.f);
                if (ACT == ACT_TANH)    x = tanhf(x);
                if (ACT == ACT_LEAKY01) x = (x > 0.f) ? x : 0.1f * x;
                C[(size_t)row * N + col] = x;
            }
        }
    }
}

// ---------------------------------------------------------------------------
// Embedding gathers (h_s/a_s go straight to d_out; they are also GEMM inputs)
// ---------------------------------------------------------------------------
__global__ void gather_embed(const float* __restrict__ team, const float* __restrict__ league,
                             const int* __restrict__ hid, const int* __restrict__ aid,
                             const int* __restrict__ lid,
                             float* __restrict__ hs, float* __restrict__ as_,
                             float* __restrict__ ls)
{
    int g = blockIdx.x * 256 + threadIdx.x;
    if (g < BATCH * DIM) {
        int b = g >> 8, d = g & 255;
        hs[g] = team[(size_t)hid[b] * DIM + d];
        as_[g] = team[(size_t)aid[b] * DIM + d];
    }
    if (g < BATCH * LDIM) {
        int b = g >> 5, d = g & 31;
        ls[g] = league[lid[b] * LDIM + d];
    }
}

// ---------------------------------------------------------------------------
// GAT: per-team scores s1 = Wh@a1, s2 = Wh@a2
// ---------------------------------------------------------------------------
__global__ __launch_bounds__(256) void gat_dots(const float* __restrict__ Wh,
                                                const float* __restrict__ a1, const float* __restrict__ a2,
                                                float* __restrict__ s1, float* __restrict__ s2)
{
    __shared__ float r1[256], r2[256];
    int team = blockIdx.x, t = threadIdx.x;
    float w = Wh[(size_t)team * DIM + t];
    r1[t] = w * a1[t]; r2[t] = w * a2[t];
    __syncthreads();
    for (int s = 128; s > 0; s >>= 1) {
        if (t < s) { r1[t] += r1[t + s]; r2[t] += r2[t + s]; }
        __syncthreads();
    }
    if (!t) { s1[team] = r1[0]; s2[team] = r2[0]; }
}

// ---------------------------------------------------------------------------
// GAT aggregation for one requested row: out[b] = softmax_j(masked lrelu(s1[j]+s2[id])) @ Wh
// ---------------------------------------------------------------------------
__global__ __launch_bounds__(256) void gat_agg(const float* __restrict__ Wh,
                                               const float* __restrict__ s1, const float* __restrict__ s2,
                                               const int* __restrict__ ids, const int* __restrict__ adj,
                                               float* __restrict__ out)
{
    __shared__ float rm[256], rs[256], wbuf[256];
    int b = blockIdx.x, t = threadIdx.x;
    int id = ids[b];
    const int* arow = adj + (size_t)id * NTEAMS;
    float s2v = s2[id];

    float m = -3.4e38f, ssum = 0.f;
    for (int j = t; j < NTEAMS; j += 256) {
        float pre = s1[j] + s2v;
        float v = (arow[j] > 0) ? (pre > 0.f ? pre : 0.2f * pre) : -9.0e15f;
        float mn = fmaxf(m, v);
        ssum = ssum * __expf(m - mn) + __expf(v - mn);
        m = mn;
    }
    rm[t] = m; rs[t] = ssum;
    __syncthreads();
    for (int s = 128; s > 0; s >>= 1) {
        if (t < s) {
            float m2 = fmaxf(rm[t], rm[t + s]);
            rs[t] = rs[t] * __expf(rm[t] - m2) + rs[t + s] * __expf(rm[t + s] - m2);
            rm[t] = m2;
        }
        __syncthreads();
    }
    float M = rm[0], invS = 1.f / rs[0];
    __syncthreads();

    float acc = 0.f;
    for (int j0 = 0; j0 < NTEAMS; j0 += 256) {
        int j = j0 + t;
        float pre = s1[j] + s2v;
        float v = (arow[j] > 0) ? (pre > 0.f ? pre : 0.2f * pre) : -9.0e15f;
        wbuf[t] = __expf(v - M) * invS;
        __syncthreads();
#pragma unroll 8
        for (int jj = 0; jj < 256; ++jj)
            acc += wbuf[jj] * Wh[(size_t)(j0 + jj) * DIM + t];
        __syncthreads();
    }
    out[(size_t)b * DIM + t] = acc;
}

// ---------------------------------------------------------------------------
// Encoder input projection + positional embedding (K=20, too small for WMMA)
// ---------------------------------------------------------------------------
__global__ void enc_input(const float* __restrict__ hseq, const float* __restrict__ aseq,
                          const float* __restrict__ inw, const float* __restrict__ inb,
                          const float* __restrict__ pos, float* __restrict__ X)
{
    int g = blockIdx.x * 256 + threadIdx.x;      // 2048*10*256
    if (g >= 2048 * SEQ * DIM) return;
    int d = g & 255;
    int s = (g >> 8) % SEQ;
    int b = g / (SEQ * DIM);
    const float* seq = (b < BATCH) ? (hseq + ((size_t)b * SEQ + s) * 20)
                                   : (aseq + ((size_t)(b - BATCH) * SEQ + s) * 20);
    float acc = inb[d] + pos[s * DIM + d];
#pragma unroll
    for (int k = 0; k < 20; ++k) acc += seq[k] * inw[k * DIM + d];
    X[g] = acc;
}

// ---------------------------------------------------------------------------
// 10x10 multi-head attention core (one block per sequence)
// ---------------------------------------------------------------------------
__global__ __launch_bounds__(256) void attn_small(const float* __restrict__ Q, const float* __restrict__ K,
                                                  const float* __restrict__ V, float* __restrict__ O)
{
    __shared__ float q[SEQ * DIM], k[SEQ * DIM], v[SEQ * DIM], sc[HEADS * SEQ * SEQ];
    int b = blockIdx.x, t = threadIdx.x;
    size_t base = (size_t)b * SEQ * DIM;
    for (int i = t; i < SEQ * DIM; i += 256) { q[i] = Q[base + i]; k[i] = K[base + i]; v[i] = V[base + i]; }
    __syncthreads();

    const float scale = 0.17677669529663687f;    // 1/sqrt(32)
    for (int p = t; p < HEADS * SEQ * SEQ; p += 256) {
        int h = p / (SEQ * SEQ), r = p % (SEQ * SEQ), i = r / SEQ, j = r % SEQ;
        const float* qp = &q[i * DIM + h * 32];
        const float* kp = &k[j * DIM + h * 32];
        float s = 0.f;
#pragma unroll
        for (int d = 0; d < 32; ++d) s += qp[d] * kp[d];
        sc[p] = s * scale;
    }
    __syncthreads();
    if (t < HEADS * SEQ) {                        // softmax over j per (h,i)
        float* row = &sc[t * SEQ];
        float m = row[0];
        for (int j = 1; j < SEQ; ++j) m = fmaxf(m, row[j]);
        float s = 0.f;
        for (int j = 0; j < SEQ; ++j) { row[j] = __expf(row[j] - m); s += row[j]; }
        float inv = 1.f / s;
        for (int j = 0; j < SEQ; ++j) row[j] *= inv;
    }
    __syncthreads();
    for (int p = t; p < SEQ * DIM; p += 256) {
        int i = p >> 8, d = p & 255, h = d >> 5;
        const float* row = &sc[(h * SEQ + i) * SEQ];
        float acc = 0.f;
#pragma unroll
        for (int j = 0; j < SEQ; ++j) acc += row[j] * v[j * DIM + d];
        O[base + p] = acc;
    }
}

// ---------------------------------------------------------------------------
// out = LayerNorm(x + r) * g + b, D=256 (in-place safe: out may alias x)
// ---------------------------------------------------------------------------
__global__ __launch_bounds__(256) void add_ln256(float* __restrict__ out, const float* __restrict__ x,
                                                 const float* __restrict__ r,
                                                 const float* __restrict__ g, const float* __restrict__ bb)
{
    __shared__ float red[256];
    int row = blockIdx.x, t = threadIdx.x;
    size_t o = (size_t)row * DIM + t;
    float val = x[o] + r[o];
    red[t] = val; __syncthreads();
    for (int s = 128; s > 0; s >>= 1) { if (t < s) red[t] += red[t + s]; __syncthreads(); }
    float mean = red[0] * (1.f / 256.f);
    __syncthreads();
    float dv = val - mean;
    red[t] = dv * dv; __syncthreads();
    for (int s = 128; s > 0; s >>= 1) { if (t < s) red[t] += red[t + s]; __syncthreads(); }
    float var = red[0] * (1.f / 256.f);
    out[o] = dv * rsqrtf(var + 1e-5f) * g[t] + bb[t];
}

// ---------------------------------------------------------------------------
// Attention pooling: u[s] = tanh-proj row . pw2 + pb2 ; w = softmax_s(u); ENC = sum_s w*x
// ---------------------------------------------------------------------------
__global__ __launch_bounds__(256) void pool_reduce(const float* __restrict__ T /*20480x128*/,
                                                   const float* __restrict__ pw2, const float* __restrict__ pb2,
                                                   const float* __restrict__ X, float* __restrict__ ENC)
{
    __shared__ float red[128], u[SEQ], w[SEQ];
    int b = blockIdx.x, t = threadIdx.x;
    for (int s = 0; s < SEQ; ++s) {
        if (t < 128) red[t] = T[((size_t)b * SEQ + s) * 128 + t] * pw2[t];
        __syncthreads();
        for (int k = 64; k > 0; k >>= 1) { if (t < k) red[t] += red[t + k]; __syncthreads(); }
        if (!t) u[s] = red[0] + pb2[0];
        __syncthreads();
    }
    if (!t) {
        float m = u[0];
        for (int s = 1; s < SEQ; ++s) m = fmaxf(m, u[s]);
        float ss = 0.f;
        for (int s = 0; s < SEQ; ++s) { w[s] = __expf(u[s] - m); ss += w[s]; }
        float inv = 1.f / ss;
        for (int s = 0; s < SEQ; ++s) w[s] *= inv;
    }
    __syncthreads();
    float acc = 0.f;
#pragma unroll
    for (int s = 0; s < SEQ; ++s) acc += w[s] * X[((size_t)b * SEQ + s) * DIM + t];
    ENC[(size_t)b * DIM + t] = acc;
}

// ---------------------------------------------------------------------------
// Gated residual network on tabular features (tiny), with LN over 32
// ---------------------------------------------------------------------------
__global__ __launch_bounds__(64) void grn_kernel(const float* __restrict__ odds,
                                                 const float* __restrict__ helo, const float* __restrict__ aelo,
                                                 const float* __restrict__ w1, const float* __restrict__ b1,
                                                 const float* __restrict__ w2, const float* __restrict__ b2,
                                                 const float* __restrict__ wg, const float* __restrict__ bg,
                                                 const float* __restrict__ wr, const float* __restrict__ br,
                                                 const float* __restrict__ ng, const float* __restrict__ nb,
                                                 float* __restrict__ tab)
{
    __shared__ float tx[5], h1[64], val[32], red[32];
    int b = blockIdx.x, t = threadIdx.x;
    if (t < 3) tx[t] = odds[b * 3 + t];
    if (t == 3) tx[3] = helo[b];
    if (t == 4) tx[4] = aelo[b];
    __syncthreads();
    float z = b1[t];
#pragma unroll
    for (int k = 0; k < 5; ++k) z += tx[k] * w1[k * 64 + t];
    h1[t] = 0.5f * z * (1.f + erff(z * 0.70710678118f));     // exact gelu
    __syncthreads();
    if (t < 32) {
        float hh = b2[t], g = bg[t], r = br[t];
        for (int k = 0; k < 64; ++k) hh += h1[k] * w2[k * 32 + t];
#pragma unroll
        for (int k = 0; k < 5; ++k) { g += tx[k] * wg[k * 32 + t]; r += tx[k] * wr[k * 32 + t]; }
        g = 1.f / (1.f + __expf(-g));
        val[t] = r + hh * g;
        red[t] = val[t];
    }
    __syncthreads();
    for (int s = 16; s > 0; s >>= 1) { if (t < s) red[t] += red[t + s]; __syncthreads(); }
    float mean = red[0] * (1.f / 32.f);
    __syncthreads();
    if (t < 32) { float d = val[t] - mean; red[t] = d * d; }
    __syncthreads();
    for (int s = 16; s > 0; s >>= 1) { if (t < s) red[t] += red[t + s]; __syncthreads(); }
    if (t < 32) {
        float var = red[0] * (1.f / 32.f);
        tab[(size_t)b * 32 + t] = (val[t] - mean) * rsqrtf(var + 1e-5f) * ng[t] + nb[t];
    }
}

// ---------------------------------------------------------------------------
// Build fusion vector (B, 2112)
// ---------------------------------------------------------------------------
__global__ void concat_fin(const float* hs, const float* as_, const float* hr, const float* ar,
                           const float* hf, const float* af, const float* ls, const float* hi,
                           const float* ai, const float* tab, float* __restrict__ fin)
{
    int g = blockIdx.x * 256 + threadIdx.x;
    if (g >= BATCH * FUSION) return;
    int b = g / FUSION, c = g % FUSION;
    float v;
    if      (c < 256)  v = hs[b * 256 + c];
    else if (c < 512)  v = as_[b * 256 + (c - 256)];
    else if (c < 768)  v = hr[b * 256 + (c - 512)];
    else if (c < 1024) v = ar[b * 256 + (c - 768)];
    else if (c < 1280) v = hf[b * 256 + (c - 1024)];
    else if (c < 1536) v = af[b * 256 + (c - 1280)];
    else if (c < 1568) v = ls[b * 32 + (c - 1536)];
    else if (c < 1824) v = hi[b * 256 + (c - 1568)];
    else if (c < 2080) v = ai[b * 256 + (c - 1824)];
    else               v = tab[b * 32 + (c - 2080)];
    fin[g] = v;
}

// ---------------------------------------------------------------------------
// Output heads: goal = s2@gw+gb ; xg = relu(s2@xw1+xb1)@xw2+xb2 (xgh precomputed)
// writes lambdas (softplus), rho (tanh*0.9), xg into d_out
// ---------------------------------------------------------------------------
__global__ __launch_bounds__(256) void finalize(const float* __restrict__ S2, const float* __restrict__ XGH,
                                                const float* __restrict__ gw, const float* __restrict__ gb,
                                                const float* __restrict__ xw2, const float* __restrict__ xb2,
                                                float* __restrict__ out)
{
    __shared__ float red[256];
    int b = blockIdx.x, t = threadIdx.x;
    float s2v = S2[(size_t)b * 256 + t];
    float goal[3];
    for (int c = 0; c < 3; ++c) {
        red[t] = s2v * gw[t * 3 + c]; __syncthreads();
        for (int s = 128; s > 0; s >>= 1) { if (t < s) red[t] += red[t + s]; __syncthreads(); }
        goal[c] = red[0] + gb[c]; __syncthreads();
    }
    float xgv[2];
    for (int c = 0; c < 2; ++c) {
        red[t] = (t < 128) ? XGH[(size_t)b * 128 + t] * xw2[t * 2 + c] : 0.f; __syncthreads();
        for (int s = 128; s > 0; s >>= 1) { if (t < s) red[t] += red[t + s]; __syncthreads(); }
        xgv[c] = red[0] + xb2[c]; __syncthreads();
    }
    if (t < 2) out[OUT_LAMBDA + b * 2 + t] = (goal[t] > 20.f) ? goal[t] : log1pf(__expf(goal[t]));
    if (t == 2) out[OUT_RHO + b] = tanhf(goal[2]) * 0.9f;
    if (t < 2) out[OUT_XG + b * 2 + t] = xgv[t];
}

// ---------------------------------------------------------------------------
// Host-side orchestration
// ---------------------------------------------------------------------------
static void launch_gemm(int act, const float* A, const float* W, const float* bias, float* C,
                        int M, int N, int K, hipStream_t s)
{
    dim3 g(N / 128, M / 128), b(256);
    switch (act) {
    case ACT_RELU:    gemm_wmma<ACT_RELU>   <<<g, b, 0, s>>>(A, W, bias, C, M, N, K); break;
    case ACT_TANH:    gemm_wmma<ACT_TANH>   <<<g, b, 0, s>>>(A, W, bias, C, M, N, K); break;
    case ACT_LEAKY01: gemm_wmma<ACT_LEAKY01><<<g, b, 0, s>>>(A, W, bias, C, M, N, K); break;
    default:          gemm_wmma<ACT_NONE>   <<<g, b, 0, s>>>(A, W, bias, C, M, N, K); break;
    }
}

// flattened input indices (setup_inputs() insertion order, params nested in order)
enum {
    IN_HSEQ = 0, IN_ASEQ, IN_HID, IN_AID, IN_LID, IN_ODDS, IN_HELO, IN_AELO, IN_ADJ,
    P_TEAM = 9, P_LEAGUE, P_GATW, P_GATA1, P_GATA2,
    P_ENC_INW = 14, P_ENC_INB, P_ENC_POS,
    P_L0 = 17,                 // per layer: wq wk wv bq bk bv wo bo w1 b1 w2 b2 n1g n1b n2g n2b (16)
    P_PW1 = 65, P_PB1, P_PW2, P_PB2,
    P_X_WQ = 69, P_X_WK, P_X_WV, P_X_BQ, P_X_BK, P_X_BV, P_X_WO, P_X_BO,
    P_NORMG = 77, P_NORMB,
    P_T_WQ = 79, P_T_WK, P_T_WV, P_T_BQ, P_T_BK, P_T_BV, P_T_WO, P_T_BO,
    P_T_N1G = 87, P_T_N1B, P_T_N2G, P_T_N2B,
    P_G_W1 = 91, P_G_B1, P_G_W2, P_G_B2, P_G_WG, P_G_BG, P_G_WR, P_G_BR, P_G_NG, P_G_NB,
    P_SW1 = 101, P_SB1, P_SW2, P_SB2, P_GW, P_GB,
    P_XW1 = 107, P_XB1, P_XW2, P_XB2
};

extern "C" void kernel_launch(void* const* d_in, const int* in_sizes, int n_in,
                              void* d_out, int out_size, void* d_ws, size_t ws_size,
                              hipStream_t stream)
{
    (void)in_sizes; (void)n_in; (void)out_size; (void)ws_size;
    auto F = [&](int i) { return (const float*)d_in[i]; };
    auto I = [&](int i) { return (const int*)d_in[i]; };
    float* out = (float*)d_out;
    float* ws  = (float*)d_ws;

    // workspace arena (floats)
    const size_t TOK = 2048ull * SEQ;             // 20480 tokens (h batch then a batch)
    float* X   = ws;                              // TOK*256
    float* Qb  = X   + TOK * 256;                 // TOK*256
    float* Kb  = Qb  + TOK * 256;                 // TOK*256
    float* Vb  = Kb  + TOK * 256;                 // TOK*256
    float* T1  = Vb  + TOK * 256;                 // TOK*256
    float* H1  = T1  + TOK * 256;                 // TOK*1024
    float* Wh  = H1  + TOK * 1024;                // 8192*256
    float* s1  = Wh  + (size_t)NTEAMS * 256;      // 8192
    float* s2  = s1  + NTEAMS;                    // 8192
    float* hr  = s2  + NTEAMS;                    // 1024*256
    float* ar  = hr  + 1024ull * 256;
    float* ls  = ar  + 1024ull * 256;             // 1024*32
    float* ENC = ls  + 1024ull * 32;              // 2048*256
    float* Vx  = ENC + 2048ull * 256;
    float* Ox  = Vx  + 2048ull * 256;
    float* hf  = Ox  + 2048ull * 256;             // 1024*256
    float* af  = hf  + 1024ull * 256;
    float* V2  = af  + 1024ull * 256;             // 2048*256
    float* O2  = V2  + 2048ull * 256;
    float* hi  = O2  + 2048ull * 256;             // 1024*256
    float* ai  = hi  + 1024ull * 256;
    float* tab = ai  + 1024ull * 256;             // 1024*32
    float* fin = tab + 1024ull * 32;              // 1024*2112
    float* S1b = fin + 1024ull * FUSION;          // 1024*512
    float* S2b = S1b + 1024ull * 512;             // 1024*256
    float* XGH = S2b + 1024ull * 256;             // 1024*128

    float* hs = out + OUT_HS;
    float* as_ = out + OUT_AS;

    // 1) embeddings (h_s/a_s straight into d_out; they're adjacent => 2048x256 GEMM input later)
    gather_embed<<<(BATCH * DIM + 255) / 256, 256, 0, stream>>>(
        F(P_TEAM), F(P_LEAGUE), I(IN_HID), I(IN_AID), I(IN_LID), hs, as_, ls);

    // 2) GAT: Wh = team_emb @ gat_W, then per-needed-row masked-softmax aggregation
    launch_gemm(ACT_NONE, F(P_TEAM), F(P_GATW), nullptr, Wh, NTEAMS, 256, 256, stream);
    gat_dots<<<NTEAMS, 256, 0, stream>>>(Wh, F(P_GATA1), F(P_GATA2), s1, s2);
    gat_agg<<<BATCH, 256, 0, stream>>>(Wh, s1, s2, I(IN_HID), I(IN_ADJ), hr);
    gat_agg<<<BATCH, 256, 0, stream>>>(Wh, s1, s2, I(IN_AID), I(IN_ADJ), ar);

    // 3) encoder input projection (h and a batched: 2048 sequences)
    enc_input<<<(2048 * SEQ * DIM + 255) / 256, 256, 0, stream>>>(
        F(IN_HSEQ), F(IN_ASEQ), F(P_ENC_INW), F(P_ENC_INB), F(P_ENC_POS), X);

    // 4) 3 transformer layers
    for (int L = 0; L < 3; ++L) {
        int p = P_L0 + 16 * L;
        launch_gemm(ACT_NONE, X, F(p + 0), F(p + 3), Qb, (int)TOK, 256, 256, stream);   // q
        launch_gemm(ACT_NONE, X, F(p + 1), F(p + 4), Kb, (int)TOK, 256, 256, stream);   // k
        launch_gemm(ACT_NONE, X, F(p + 2), F(p + 5), Vb, (int)TOK, 256, 256, stream);   // v
        attn_small<<<2048, 256, 0, stream>>>(Qb, Kb, Vb, T1);
        launch_gemm(ACT_NONE, T1, F(p + 6), F(p + 7), Qb, (int)TOK, 256, 256, stream);  // o-proj
        add_ln256<<<(int)TOK, 256, 0, stream>>>(X, X, Qb, F(p + 12), F(p + 13));
        launch_gemm(ACT_RELU, X, F(p + 8), F(p + 9), H1, (int)TOK, 1024, 256, stream);  // ffn up
        launch_gemm(ACT_NONE, H1, F(p + 10), F(p + 11), T1, (int)TOK, 256, 1024, stream); // ffn down
        add_ln256<<<(int)TOK, 256, 0, stream>>>(X, X, T1, F(p + 14), F(p + 15));
    }

    // 5) attention pooling -> ENC (h_c rows 0..1023, a_c rows 1024..2047)
    launch_gemm(ACT_TANH, X, F(P_PW1), F(P_PB1), Qb, (int)TOK, 128, 256, stream);
    pool_reduce<<<2048, 256, 0, stream>>>(Qb, F(P_PW2), F(P_PB2), X, ENC);

    // 6) cross attention with L=1 collapses to (v@wv+bv)@wo+bo with swapped sides
    launch_gemm(ACT_NONE, ENC, F(P_X_WV), F(P_X_BV), Vx, 2048, 256, 256, stream);
    launch_gemm(ACT_NONE, Vx, F(P_X_WO), F(P_X_BO), Ox, 2048, 256, 256, stream);
    add_ln256<<<BATCH, 256, 0, stream>>>(hf, ENC, Ox + 1024ull * 256, F(P_NORMG), F(P_NORMB));
    add_ln256<<<BATCH, 256, 0, stream>>>(af, ENC + 1024ull * 256, Ox, F(P_NORMG), F(P_NORMB));

    // 7) tactical block (also L=1 collapse); input = [h_s; a_s] contiguous in d_out
    launch_gemm(ACT_NONE, hs, F(P_T_WV), F(P_T_BV), V2, 2048, 256, 256, stream);
    launch_gemm(ACT_NONE, V2, F(P_T_WO), F(P_T_BO), O2, 2048, 256, 256, stream);
    add_ln256<<<BATCH, 256, 0, stream>>>(hi, hs, O2 + 1024ull * 256, F(P_T_N1G), F(P_T_N1B));
    add_ln256<<<BATCH, 256, 0, stream>>>(ai, as_, O2, F(P_T_N2G), F(P_T_N2B));

    // 8) gated residual network on tabular features
    grn_kernel<<<BATCH, 64, 0, stream>>>(F(IN_ODDS), F(IN_HELO), F(IN_AELO),
                                         F(P_G_W1), F(P_G_B1), F(P_G_W2), F(P_G_B2),
                                         F(P_G_WG), F(P_G_BG), F(P_G_WR), F(P_G_BR),
                                         F(P_G_NG), F(P_G_NB), tab);

    // 9) fusion MLP + heads
    concat_fin<<<(BATCH * FUSION + 255) / 256, 256, 0, stream>>>(hs, as_, hr, ar, hf, af, ls, hi, ai, tab, fin);
    launch_gemm(ACT_LEAKY01, fin, F(P_SW1), F(P_SB1), S1b, BATCH, 512, FUSION, stream);
    launch_gemm(ACT_LEAKY01, S1b, F(P_SW2), F(P_SB2), S2b, BATCH, 256, 512, stream);
    launch_gemm(ACT_RELU, S2b, F(P_XW1), F(P_XB1), XGH, BATCH, 128, 256, stream);
    finalize<<<BATCH, 256, 0, stream>>>(S2b, XGH, F(P_GW), F(P_GB), F(P_XW2), F(P_XB2), out);
}
